// InterventionImpactLayer_82025285419548
// MI455X (gfx1250) — matrix operand order repaired and checked by reading, not compiled, not run on hardware
//
#include <hip/hip_runtime.h>
#include <hip/hip_bf16.h>

// ---------------------------------------------------------------------------
// MI455X (gfx1250) implementation.
// All dense GEMMs (encoder, GAT projection, impact head, LSTM gates) run on
// v_wmma_f32_16x16x32_bf16. Weights are pre-converted ONCE per launch into
// bf16 [col][K] (transposed, K-padded) so B tiles are contiguous packed-pair
// dword loads with no per-wave conversion. Activations that feed GEMMs are
// kept as bf16 mirrors, so A tiles are packed dword loads too. The only
// remaining f32->bf16 conversion in hot GEMM loops is the encoder's A (raw
// input x). Irregular edge gather/scatter uses global f32 atomics
// (51 MB arrays -> L2-resident). wave32, full EXEC wherever WMMA runs
// (N = 6250 * 16 exactly).
// ---------------------------------------------------------------------------

#define NN 100000
#define EE 1600000

typedef __bf16 bf16_t;
typedef __attribute__((ext_vector_type(16))) __bf16 v16bf;
typedef __attribute__((ext_vector_type(8)))  float  v8f;

__device__ inline bf16_t f2bf(float f) { return (bf16_t)f; }  // v_cvt_pk_bf16_f32

__device__ inline v8f wmma_bf16(v16bf a, v16bf b, v8f c) {
  return __builtin_amdgcn_wmma_f32_16x16x32_bf16(
      /*neg_a=*/false, a, /*neg_b=*/false, b,
      /*c_mod=*/(short)0, c, /*reuse_a=*/false, /*reuse_b=*/false);
}

// ---- A tile from bf16 row-major [rows][lda]: 16 rows x 32 K ---------------
// ISA A layout: lane half=lane>>4, m=lane&15;
// VGPR v holds K pair k = 16*(v>>2) + 8*half + 2*(v&3); dword idx = k/2.
__device__ inline v16bf load_a_bf(const bf16_t* __restrict__ A, int lda,
                                  int row0, int k0, int lane) {
  int half = lane >> 4, m = lane & 15;
  const unsigned* rp = (const unsigned*)(A + (size_t)(row0 + m) * lda + k0);
  union { v16bf v; unsigned u[8]; } a;
#pragma unroll
  for (int v = 0; v < 8; ++v)
    a.u[v] = rp[((v >> 2) << 3) + (half << 2) + (v & 3)];
  return a.v;
}

// ---- B tile from bf16 transposed weights Wt[col][K] (ldw = K) -------------
// ISA B layout: lane n=lane&15, half=lane>>4; VGPR v holds K pair 16*(v>>2) + 8*half + 2*(v&3).
__device__ inline v16bf load_bt_bf(const bf16_t* __restrict__ Wt, int ldw,
                                   int col0, int k0, int lane) {
  int half = lane >> 4, n = lane & 15;
  const unsigned* rp = (const unsigned*)(Wt + (size_t)(col0 + n) * ldw + k0);
  union { v16bf v; unsigned u[8]; } b;
#pragma unroll
  for (int v = 0; v < 8; ++v)
    b.u[v] = rp[((v >> 2) << 3) + (half << 2) + (v & 3)];
  return b.v;
}

// ---- Encoder A = concat(x[N,128], itype[N,3]) zero-padded to K=160 (f32) --
__device__ inline v16bf load_a_enc(const float* __restrict__ x,
                                   const float* __restrict__ it,
                                   int row0, int k0, int lane) {
  int half = lane >> 4, m = lane & 15;
  int row = row0 + m;
  v16bf a;
#pragma unroll
  for (int v = 0; v < 8; ++v) {
    int k = k0 + ((v >> 2) << 4) + (half << 3) + ((v & 3) << 1);
#pragma unroll
    for (int j = 0; j < 2; ++j) {
      int kk = k + j;
      float f = (kk < 128) ? x[(size_t)row * 128 + kk]
                           : (kk < 131 ? it[(size_t)row * 3 + (kk - 128)] : 0.f);
      a[2 * v + j] = f2bf(f);
    }
  }
  return a;
}

__device__ inline void atomicMaxFloat(float* addr, float val) {
  if (val >= 0.f) atomicMax((int*)addr, __float_as_int(val));
  else            atomicMin((unsigned int*)addr, __float_as_uint(val));
}

__device__ inline float sigmoidf(float x) { return 1.f / (1.f + expf(-x)); }

// ---------------- one-time weight conversion kernels -----------------------
// fp32 row-major [K][C] -> bf16 transposed [C][Kpad], zero pad k >= K
__global__ void convert_wt_kernel(const float* __restrict__ in,
                                  bf16_t* __restrict__ out,
                                  int K, int C, int Kpad) {
  int i = blockIdx.x * 256 + threadIdx.x;       // c*Kpad + k
  if (i >= C * Kpad) return;
  int c = i / Kpad, k = i - c * Kpad;
  out[i] = (k < K) ? f2bf(in[(size_t)k * C + c]) : f2bf(0.f);
}

// plain elementwise fp32 -> bf16 (layout already [col][K])
__global__ void convert_w_kernel(const float* __restrict__ in,
                                 bf16_t* __restrict__ out, int n) {
  int i = blockIdx.x * 256 + threadIdx.x;
  if (i >= n) return;
  out[i] = f2bf(in[i]);
}

// activation mirror fp32 -> bf16
__global__ void convert_act_kernel(const float* __restrict__ in,
                                   bf16_t* __restrict__ out) {
  size_t i = (size_t)blockIdx.x * 256 + threadIdx.x;
  if (i >= (size_t)NN * 128) return;
  out[i] = f2bf(in[i]);
}

// ---------------- encoder: h = relu([x|itype] @ W_enc + b) * mask ----------
// One wave per 16-row strip; 8 column tiles; B = pre-transposed bf16 [128][160].
__global__ void encoder_kernel(const float* __restrict__ x,
                               const float* __restrict__ itype,
                               const bf16_t* __restrict__ Wt,
                               const float* __restrict__ bias,
                               const float* __restrict__ mask,
                               bf16_t* __restrict__ outb) {
  int wid  = (blockIdx.x * blockDim.x + threadIdx.x) >> 5;
  int lane = threadIdx.x & 31;
  int row0 = wid * 16;
  v8f acc[8] = {};
#pragma unroll
  for (int k0 = 0; k0 < 160; k0 += 32) {
    v16bf a = load_a_enc(x, itype, row0, k0, lane);
#pragma unroll
    for (int ct = 0; ct < 8; ++ct) {
      v16bf b = load_bt_bf(Wt, 160, ct * 16, k0, lane);
      acc[ct] = wmma_bf16(a, b, acc[ct]);
    }
  }
  int half = lane >> 4, n = lane & 15;
#pragma unroll
  for (int ct = 0; ct < 8; ++ct) {
    int col = ct * 16 + n;
    float bv = bias[col];
#pragma unroll
    for (int i = 0; i < 8; ++i) {
      int row = row0 + half * 8 + i;
      float v = acc[ct][i] + bv;
      v = v > 0.f ? v : 0.f;
      outb[(size_t)row * 128 + col] = f2bf(v * mask[row]);
    }
  }
}

// ------- strip GEMM, K=128: C[N, CT*16] = act(A_bf16 @ Wt_bf16 (+bias)) ----
template <int CT, bool RELU, bool HASBIAS>
__global__ void gemm_bf_strip_kernel(const bf16_t* __restrict__ A,
                                     const bf16_t* __restrict__ Wt,
                                     const float* __restrict__ bias,
                                     float* __restrict__ C) {
  int wid  = (blockIdx.x * blockDim.x + threadIdx.x) >> 5;
  int lane = threadIdx.x & 31;
  int row0 = wid * 16;
  const int ncols = CT * 16;
  v8f acc[CT] = {};
#pragma unroll
  for (int k0 = 0; k0 < 128; k0 += 32) {
    v16bf a = load_a_bf(A, 128, row0, k0, lane);
#pragma unroll
    for (int ct = 0; ct < CT; ++ct) {
      v16bf b = load_bt_bf(Wt, 128, ct * 16, k0, lane);
      acc[ct] = wmma_bf16(a, b, acc[ct]);
    }
  }
  int half = lane >> 4, n = lane & 15;
#pragma unroll
  for (int ct = 0; ct < CT; ++ct) {
    int col = ct * 16 + n;
    float bv = HASBIAS ? bias[col] : 0.f;
#pragma unroll
    for (int i = 0; i < 8; ++i) {
      float v = acc[ct][i] + bv;
      if (RELU) v = v > 0.f ? v : 0.f;
      C[(size_t)(row0 + half * 8 + i) * ncols + col] = v;
    }
  }
}

// ---------------- GAT attention prep ---------------------------------------
__global__ void attn_prep_kernel(const float* __restrict__ xh,
                                 const float* __restrict__ att_s,
                                 const float* __restrict__ att_d,
                                 float* __restrict__ a_s, float* __restrict__ a_d,
                                 float* __restrict__ m, float* __restrict__ s) {
  int i = blockIdx.x * blockDim.x + threadIdx.x;   // node*4 + head
  if (i >= NN * 4) return;
  int node = i >> 2, head = i & 3;
  const float* xp = xh + (size_t)node * 128 + head * 32;
  const float* as = att_s + head * 32;
  const float* ad = att_d + head * 32;
  float vs = 0.f, vd = 0.f;
#pragma unroll
  for (int j = 0; j < 32; ++j) { float v = xp[j]; vs += v * as[j]; vd += v * ad[j]; }
  a_s[i] = vs; a_d[i] = vd; m[i] = -1e30f; s[i] = 0.f;
}

__global__ void init_out_kernel(float* __restrict__ out, const float* __restrict__ bias) {
  size_t i = (size_t)blockIdx.x * blockDim.x + threadIdx.x;
  if (i >= (size_t)NN * 128) return;
  out[i] = bias[i & 127];
}

// ---------------- edge kernels (E' = E + N self-loops) ---------------------
__device__ inline void edge_pair(const int* __restrict__ ei, int e, int& src, int& dst) {
  if (e < EE) { src = ei[e]; dst = ei[EE + e]; }
  else        { src = e - EE; dst = e - EE; }
}

__global__ void edge_max_kernel(const int* __restrict__ ei,
                                const float* __restrict__ a_s,
                                const float* __restrict__ a_d,
                                float* __restrict__ m) {
  int e = blockIdx.x * blockDim.x + threadIdx.x;
  if (e >= EE + NN) return;
  int src, dst; edge_pair(ei, e, src, dst);
#pragma unroll
  for (int h = 0; h < 4; ++h) {
    float v = a_s[src * 4 + h] + a_d[dst * 4 + h];
    v = v > 0.f ? v : 0.2f * v;
    atomicMaxFloat(&m[dst * 4 + h], v);
  }
}

__global__ void edge_sum_kernel(const int* __restrict__ ei,
                                const float* __restrict__ a_s,
                                const float* __restrict__ a_d,
                                const float* __restrict__ m,
                                float* __restrict__ s) {
  int e = blockIdx.x * blockDim.x + threadIdx.x;
  if (e >= EE + NN) return;
  int src, dst; edge_pair(ei, e, src, dst);
#pragma unroll
  for (int h = 0; h < 4; ++h) {
    float v = a_s[src * 4 + h] + a_d[dst * 4 + h];
    v = v > 0.f ? v : 0.2f * v;
    atomicAdd(&s[dst * 4 + h], expf(v - m[dst * 4 + h]));
  }
}

// one wave per edge; lane l handles dims l, 32+l, 64+l, 96+l (heads 0..3)
__global__ void edge_msg_kernel(const int* __restrict__ ei,
                                const float* __restrict__ a_s,
                                const float* __restrict__ a_d,
                                const float* __restrict__ m,
                                const float* __restrict__ s,
                                const float* __restrict__ xh,
                                float* __restrict__ out) {
  int gid = blockIdx.x * blockDim.x + threadIdx.x;
  int e = gid >> 5, lane = gid & 31;
  if (e >= EE + NN) return;
  int src, dst; edge_pair(ei, e, src, dst);
  const float* xr = xh + (size_t)src * 128;
  float* orow = out + (size_t)dst * 128;
  __builtin_prefetch(xr + lane, 0, 1);            // global_prefetch_b8
#pragma unroll
  for (int h = 0; h < 4; ++h) {
    float v = a_s[src * 4 + h] + a_d[dst * 4 + h];
    v = v > 0.f ? v : 0.2f * v;
    float alpha = expf(v - m[dst * 4 + h]) / (s[dst * 4 + h] + 1e-16f);
    int d = h * 32 + lane;
    atomicAdd(&orow[d], xr[d] * alpha);
  }
}

// ---------------- impact head part 2: tmp[N,64] @ W2[64,3] + b2 ------------
__global__ void impact2_kernel(const float* __restrict__ tmp,
                               const float* __restrict__ W2,
                               const float* __restrict__ b2,
                               float* __restrict__ imp,
                               float* __restrict__ total, float scale) {
  __shared__ float red[256];
  int n = blockIdx.x * 256 + threadIdx.x;
  float asum = 0.f;
  if (n < NN) {
    float s0 = b2[0], s1 = b2[1], s2 = b2[2];
    const float* t = tmp + (size_t)n * 64;
#pragma unroll
    for (int j = 0; j < 64; ++j) {
      float v = t[j];
      s0 += v * W2[j * 3 + 0];
      s1 += v * W2[j * 3 + 1];
      s2 += v * W2[j * 3 + 2];
    }
    imp[(size_t)n * 3 + 0] = s0;
    imp[(size_t)n * 3 + 1] = s1;
    imp[(size_t)n * 3 + 2] = s2;
    asum = fabsf(s0) + fabsf(s1) + fabsf(s2);
  }
  red[threadIdx.x] = asum;
  __syncthreads();
  for (int off = 128; off > 0; off >>= 1) {
    if (threadIdx.x < off) red[threadIdx.x] += red[threadIdx.x + off];
    __syncthreads();
  }
  if (threadIdx.x == 0) atomicAdd(total, red[0] * scale);
}

// ---------------- fused LSTM step ------------------------------------------
// One block = 16 nodes. 8 waves x 4 col-tiles cover the 512 gate columns.
// A (bf16 mirrors) loaded once per K chunk; B from pre-converted bf16
// Wih/Whh (natively [col][K]). Gates staged in LDS; i/f/g/o nonlinearity and
// c/h update done in-block. Writes f32 h (for h_last), bf16 h mirror (next
// GEMM), and y output either as bf16 (layer 0) or f32 (layer 1 -> d_out).
__global__ void lstm_step_kernel(const bf16_t* __restrict__ xtb,
                                 const bf16_t* __restrict__ Wih,
                                 const bf16_t* __restrict__ Whh,
                                 const float* __restrict__ bih,
                                 const float* __restrict__ bhh,
                                 float* __restrict__ h_state,
                                 float* __restrict__ c_state,
                                 bf16_t* __restrict__ h_state_b,
                                 float* __restrict__ youtf, int ldyf,
                                 bf16_t* __restrict__ youtb, int ldyb) {
  __shared__ float gates[16 * 512];
  int tid = threadIdx.x;
  int wave = tid >> 5, lane = tid & 31;
  int row0 = blockIdx.x * 16;
  int half = lane >> 4, nn = lane & 15;

  v8f acc[4] = {};
  for (int k0 = 0; k0 < 256; k0 += 32) {
    const bf16_t* Asrc = (k0 < 128) ? xtb : h_state_b;
    const bf16_t* Wsrc = (k0 < 128) ? Wih : Whh;
    int kk = k0 & 127;
    v16bf a = load_a_bf(Asrc, 128, row0, kk, lane);
#pragma unroll
    for (int ct = 0; ct < 4; ++ct) {
      int col0 = wave * 64 + ct * 16;
      v16bf b = load_bt_bf(Wsrc, 128, col0, kk, lane);
      acc[ct] = wmma_bf16(a, b, acc[ct]);
    }
  }
#pragma unroll
  for (int ct = 0; ct < 4; ++ct) {
    int col = wave * 64 + ct * 16 + nn;
    float bsum = bih[col] + bhh[col];
#pragma unroll
    for (int i = 0; i < 8; ++i)
      gates[(half * 8 + i) * 512 + col] = acc[ct][i] + bsum;
  }
  __syncthreads();
#pragma unroll
  for (int it = 0; it < 8; ++it) {
    int lin = it * 256 + tid;       // 16 nodes * 128 dims
    int mrow = lin >> 7, d = lin & 127;
    int node = row0 + mrow;
    float gi = gates[mrow * 512 + d];
    float gf = gates[mrow * 512 + 128 + d];
    float gg = gates[mrow * 512 + 256 + d];
    float go = gates[mrow * 512 + 384 + d];
    float cp = c_state[(size_t)node * 128 + d];
    float cn = sigmoidf(gf) * cp + sigmoidf(gi) * tanhf(gg);
    float hn = sigmoidf(go) * tanhf(cn);
    c_state[(size_t)node * 128 + d] = cn;
    h_state[(size_t)node * 128 + d] = hn;
    h_state_b[(size_t)node * 128 + d] = f2bf(hn);
    if (youtf) youtf[(size_t)node * ldyf + d] = hn;
    if (youtb) youtb[(size_t)node * ldyb + d] = f2bf(hn);
  }
}

// ---------------------------------------------------------------------------
extern "C" void kernel_launch(void* const* d_in, const int* in_sizes, int n_in,
                              void* d_out, int out_size, void* d_ws, size_t ws_size,
                              hipStream_t stream) {
  const float* x      = (const float*)d_in[0];
  const int*   ei     = (const int*)  d_in[1];
  const float* mask   = (const float*)d_in[2];
  const float* itype  = (const float*)d_in[3];
  const float* W_enc  = (const float*)d_in[4];
  const float* b_enc  = (const float*)d_in[5];
  const float* W_gat  = (const float*)d_in[6];   // [3,128,128]
  const float* att_s  = (const float*)d_in[7];   // [3,4,32]
  const float* att_d  = (const float*)d_in[8];
  const float* b_gat  = (const float*)d_in[9];   // [3,128]
  const float* W1     = (const float*)d_in[10];  // [3,128,64]
  const float* b1     = (const float*)d_in[11];  // [3,64]
  const float* W2     = (const float*)d_in[12];  // [3,64,3]
  const float* b2     = (const float*)d_in[13];  // [3,3]
  const float* Wih0   = (const float*)d_in[14];  // [512,128]
  const float* Whh0   = (const float*)d_in[15];
  const float* bih0   = (const float*)d_in[16];
  const float* bhh0   = (const float*)d_in[17];
  const float* Wih1   = (const float*)d_in[18];
  const float* Whh1   = (const float*)d_in[19];
  const float* bih1   = (const float*)d_in[20];
  const float* bhh1   = (const float*)d_in[21];

  // ---- workspace: f32 region then bf16 region ----
  float* wsf = (float*)d_ws;
  float* hout    = wsf; wsf += (size_t)NN * 128;   // per-hop atomic accumulator
  float* xh      = wsf; wsf += (size_t)NN * 128;
  float* a_sb    = wsf; wsf += (size_t)NN * 4;
  float* a_db    = wsf; wsf += (size_t)NN * 4;
  float* mbuf    = wsf; wsf += (size_t)NN * 4;
  float* sbuf    = wsf; wsf += (size_t)NN * 4;
  float* tmp64   = wsf; wsf += (size_t)NN * 64;
  float* h_state = wsf; wsf += (size_t)NN * 128;
  float* c_state = wsf; wsf += (size_t)NN * 128;

  bf16_t* wsb = (bf16_t*)wsf;
  bf16_t* h_enc_b   = wsb; wsb += (size_t)NN * 128;
  bf16_t* hops_b    = wsb; wsb += (size_t)3 * NN * 128;
  bf16_t* y0_b      = wsb; wsb += (size_t)3 * NN * 128;
  bf16_t* h_state_b = wsb; wsb += (size_t)NN * 128;
  bf16_t* WbT_enc   = wsb; wsb += 128 * 160;       // [col][Kpad=160]
  bf16_t* WbT_gat   = wsb; wsb += 3 * 128 * 128;   // [col][128]
  bf16_t* WbT_1     = wsb; wsb += 3 * 64 * 128;    // [col][128]
  bf16_t* Wb_ih0    = wsb; wsb += 512 * 128;       // natively [col][K]
  bf16_t* Wb_hh0    = wsb; wsb += 512 * 128;
  bf16_t* Wb_ih1    = wsb; wsb += 512 * 128;
  bf16_t* Wb_hh1    = wsb; wsb += 512 * 128;

  float* out      = (float*)d_out;
  float* imp0     = out;
  float* imp1     = out + (size_t)3 * NN;
  float* imp2_    = out + (size_t)6 * NN;
  float* temporal = out + (size_t)9 * NN;          // [N,3,128]
  float* h_last   = out + (size_t)393 * NN;        // [N,128]
  float* total    = out + (size_t)521 * NN;        // scalar

  hipMemsetAsync(total, 0, sizeof(float), stream);

  // ---- one-time weight conversion / transposition ----
  convert_wt_kernel<<<(128 * 160 + 255) / 256, 256, 0, stream>>>(W_enc, WbT_enc, 131, 128, 160);
  for (int k = 0; k < 3; ++k) {
    convert_wt_kernel<<<(128 * 128 + 255) / 256, 256, 0, stream>>>(
        W_gat + (size_t)k * 128 * 128, WbT_gat + (size_t)k * 128 * 128, 128, 128, 128);
    convert_wt_kernel<<<(64 * 128 + 255) / 256, 256, 0, stream>>>(
        W1 + (size_t)k * 128 * 64, WbT_1 + (size_t)k * 64 * 128, 128, 64, 128);
  }
  convert_w_kernel<<<256, 256, 0, stream>>>(Wih0, Wb_ih0, 512 * 128);
  convert_w_kernel<<<256, 256, 0, stream>>>(Whh0, Wb_hh0, 512 * 128);
  convert_w_kernel<<<256, 256, 0, stream>>>(Wih1, Wb_ih1, 512 * 128);
  convert_w_kernel<<<256, 256, 0, stream>>>(Whh1, Wb_hh1, 512 * 128);

  const int rowWaves    = NN / 16;                 // 6250 strips, exact
  const int stripBlocks = rowWaves / 2;            // 64-thread blocks (2 waves)
  const int actBlocks   = (int)(((size_t)NN * 128 + 255) / 256);

  // encoder -> bf16 h mirror only (it only feeds the hop-0 GEMM)
  encoder_kernel<<<stripBlocks, 64, 0, stream>>>(x, itype, WbT_enc, b_enc, mask, h_enc_b);

  const int edgeBlocks = (EE + NN + 255) / 256;
  const int msgBlocks  = (EE + NN) * 32 / 256;     // exact
  float* imps[3] = {imp0, imp1, imp2_};
  const float scales[3] = {1.f / (3.f * NN), 0.5f / (3.f * NN), 0.25f / (3.f * NN)};

  for (int k = 0; k < 3; ++k) {
    const bf16_t* hin_b = (k == 0) ? h_enc_b : hops_b + (size_t)(k - 1) * NN * 128;
    bf16_t*       hb    = hops_b + (size_t)k * NN * 128;
    // xh = hin @ W_gat[k]   (128 cols, 8 tiles per wave)
    gemm_bf_strip_kernel<8, false, false><<<stripBlocks, 64, 0, stream>>>(
        hin_b, WbT_gat + (size_t)k * 128 * 128, nullptr, xh);
    attn_prep_kernel<<<(NN * 4 + 255) / 256, 256, 0, stream>>>(
        xh, att_s + k * 128, att_d + k * 128, a_sb, a_db, mbuf, sbuf);
    init_out_kernel<<<actBlocks, 256, 0, stream>>>(hout, b_gat + k * 128);
    edge_max_kernel<<<edgeBlocks, 256, 0, stream>>>(ei, a_sb, a_db, mbuf);
    edge_sum_kernel<<<edgeBlocks, 256, 0, stream>>>(ei, a_sb, a_db, mbuf, sbuf);
    edge_msg_kernel<<<msgBlocks, 256, 0, stream>>>(ei, a_sb, a_db, mbuf, sbuf, xh, hout);
    // hop output -> bf16 mirror (feeds impact GEMM, next hop, LSTM)
    convert_act_kernel<<<actBlocks, 256, 0, stream>>>(hout, hb);
    // impact head: tmp = relu(h @ W1[k] + b1[k])  (64 cols, 4 tiles/wave)
    gemm_bf_strip_kernel<4, true, true><<<stripBlocks, 64, 0, stream>>>(
        hb, WbT_1 + (size_t)k * 64 * 128, b1 + k * 64, tmp64);
    impact2_kernel<<<(NN + 255) / 256, 256, 0, stream>>>(
        tmp64, W2 + k * 64 * 3, b2 + k * 3, imps[k], total, scales[k]);
  }

  // LSTM layer 0: hops_b -> y0_b (bf16 only; it only feeds the layer-1 GEMM)
  hipMemsetAsync(h_state_b, 0, (size_t)NN * 128 * sizeof(bf16_t), stream);
  hipMemsetAsync(c_state, 0, (size_t)NN * 128 * sizeof(float), stream);
  for (int t = 0; t < 3; ++t)
    lstm_step_kernel<<<rowWaves, 256, 0, stream>>>(
        hops_b + (size_t)t * NN * 128, Wb_ih0, Wb_hh0, bih0, bhh0,
        h_state, c_state, h_state_b,
        nullptr, 0, y0_b + (size_t)t * NN * 128, 128);

  // LSTM layer 1: y0_b -> temporal [N,3,128] (f32, straight into d_out)
  hipMemsetAsync(h_state_b, 0, (size_t)NN * 128 * sizeof(bf16_t), stream);
  hipMemsetAsync(c_state, 0, (size_t)NN * 128 * sizeof(float), stream);
  for (int t = 0; t < 3; ++t)
    lstm_step_kernel<<<rowWaves, 256, 0, stream>>>(
        y0_b + (size_t)t * NN * 128, Wb_ih1, Wb_hh1, bih1, bhh1,
        h_state, c_state, h_state_b,
        temporal + (size_t)t * 128, 384, nullptr, 0);

  hipMemcpyAsync(h_last, h_state, (size_t)NN * 128 * sizeof(float),
                 hipMemcpyDeviceToDevice, stream);
}
